// LsrHierGNNLayer_43920335569327
// MI455X (gfx1250) — matrix operand order repaired and checked by reading, not compiled, not run on hardware
//
#include <hip/hip_runtime.h>
#include <hip/hip_bf16.h>
#include <math.h>

// ============================================================================
// LsrHierGNNLayer for MI455X (gfx1250, wave32, WMMA).
// All large GEMMs use v_wmma_f32_16x16x32_bf16 (bf16 in, f32 accum),
// double-buffered LDS, b128 staging, templated straight-line epilogues.
// Requires ws_size >= ~500 MiB (layout below, phase-aliased).
// B=64, S=256, H=1024, HOPS=2.
// ============================================================================

typedef __attribute__((ext_vector_type(16))) __bf16 v16bf;
typedef __attribute__((ext_vector_type(8)))  float  v8f;
typedef __bf16 bf16;

#define Bn   64
#define Sn   256
#define Hn   1024
#define EPSC 1e-6f
#define LNE  1e-5f

// ---------------------------------------------------------------------------
// Generic bf16 WMMA GEMM:  out[b] = act( A[b] (MxK) @ Bt[b]^T + bias )
// Bt is row-major [N][K].  Block tile 128x128, BK=32, 8 waves (wave32),
// wave tile 32x64 (2x4 WMMA).  LDS rows padded to 40 elements (80B):
// 16B-aligned for b128 and bank-conflict-free (20*r mod 64 distinct).
// MODE: 0 = none, 1 = relu, 2 = scores->A-matrix epilogue.
// ---------------------------------------------------------------------------
#define BM 128
#define BN 128
#define BK 32
#define LDP 40

union FragU { v16bf v; unsigned int u[8]; };

static __device__ __forceinline__ void ld4(uint4& d, const bf16* p)
{ d = *(const uint4*)p; }
static __device__ __forceinline__ void st4(bf16* p, const uint4& d)
{ *(uint4*)p = d; }

template<int MODE, bool BIASF, bool WF, bool WB>
__global__ __launch_bounds__(256) void k_gemm(
    const bf16* __restrict__ A,  int lda, size_t strideA,
    const bf16* __restrict__ Bt, int ldb, size_t strideB,
    const float* __restrict__ bias,
    float* __restrict__ outF, bf16* __restrict__ outB, int ldc, size_t strideC,
    int K, const float* __restrict__ mask, int Sdim)
{
    __shared__ bf16 As[2][BM * LDP];
    __shared__ bf16 Bs[2][BN * LDP];

    const int b = blockIdx.z;
    const int tid  = threadIdx.x;
    const int lane = tid & 31;
    const int w    = tid >> 5;
    const int wm   = w & 3;        // 0..3 -> 32-row strip
    const int wn   = w >> 2;       // 0..1 -> 64-col strip
    const int half = lane >> 4;    // 0/1
    const int l15  = lane & 15;

    const int rowBase = blockIdx.y * BM;
    const int colBase = blockIdx.x * BN;

    // staging: tile = 128 rows x 32 bf16 = 512 uint4; 2 per thread.
    // q -> row = q>>2, part = q&3 (8 elements each)
    const int q0 = tid, q1 = tid + 256;
    const int r0 = q0 >> 2, p0 = (q0 & 3) * 8;
    const int r1 = q1 >> 2, p1 = (q1 & 3) * 8;

    const bf16* Ab = A  + (size_t)b * strideA;
    const bf16* Bb = Bt + (size_t)b * strideB;
    const bf16* aP0 = Ab + (size_t)(rowBase + r0) * lda + p0;
    const bf16* aP1 = Ab + (size_t)(rowBase + r1) * lda + p1;
    const bf16* bP0 = Bb + (size_t)(colBase + r0) * ldb + p0;
    const bf16* bP1 = Bb + (size_t)(colBase + r1) * ldb + p1;
    const int sA0 = r0 * LDP + p0, sA1 = r1 * LDP + p1;

    v8f acc[2][4] = {};

    uint4 rA0, rA1, rB0, rB1;
    ld4(rA0, aP0); ld4(rA1, aP1); ld4(rB0, bP0); ld4(rB1, bP1);
    st4(&As[0][sA0], rA0); st4(&As[0][sA1], rA1);
    st4(&Bs[0][sA0], rB0); st4(&Bs[0][sA1], rB1);
    __syncthreads();

    const int nk = K / BK;
    for (int kt = 0; kt < nk; ++kt) {
        const int cur = kt & 1;
        const bool more = (kt + 1 < nk);
        if (more) {
            int k0 = (kt + 1) * BK;
            ld4(rA0, aP0 + k0); ld4(rA1, aP1 + k0);
            ld4(rB0, bP0 + k0); ld4(rB1, bP1 + k0);
        }

        const bf16* Ac = As[cur];
        const bf16* Bc = Bs[cur];

        // A fragments: lane row = l15; half=0 -> K{0..7,16..23}, half=1 -> K{8..15,24..31}
        FragU afr[2];
        #pragma unroll
        for (int mi = 0; mi < 2; ++mi) {
            int row = wm * 32 + mi * 16 + l15;
            int kb  = half * 8;
            ld4(*(uint4*)&afr[mi].u[0], &Ac[row * LDP + kb]);
            ld4(*(uint4*)&afr[mi].u[4], &Ac[row * LDP + kb + 16]);
        }
        // B fragments: lane col = l15; half=0 -> K{0..7,16..23}, half=1 -> K{8..15,24..31}
        FragU bfr[4];
        #pragma unroll
        for (int ni = 0; ni < 4; ++ni) {
            int n  = wn * 64 + ni * 16 + l15;
            int kb = half * 8;
            ld4(*(uint4*)&bfr[ni].u[0], &Bc[n * LDP + kb]);
            ld4(*(uint4*)&bfr[ni].u[4], &Bc[n * LDP + kb + 16]);
        }

        #pragma unroll
        for (int mi = 0; mi < 2; ++mi)
            #pragma unroll
            for (int ni = 0; ni < 4; ++ni)
                acc[mi][ni] = __builtin_amdgcn_wmma_f32_16x16x32_bf16(
                    false, afr[mi].v, false, bfr[ni].v,
                    (short)0, acc[mi][ni], false, false);

        if (more) {
            bf16* An = As[cur ^ 1];
            bf16* Bn_ = Bs[cur ^ 1];
            st4(&An[sA0], rA0); st4(&An[sA1], rA1);
            st4(&Bn_[sA0], rB0); st4(&Bn_[sA1], rB1);
        }
        __syncthreads();
    }

    // epilogue (C/D layout: lanes 0-15 -> M=r, lanes 16-31 -> M=r+8; N = l15)
    #pragma unroll
    for (int ni = 0; ni < 4; ++ni) {
        const int n = colBase + wn * 64 + ni * 16 + l15;
        float bv = 0.0f, mj = 0.0f;
        if constexpr (BIASF)    bv = bias[n];
        if constexpr (MODE == 2) mj = mask[(size_t)b * Sdim + n];
        #pragma unroll
        for (int mi = 0; mi < 2; ++mi) {
            #pragma unroll
            for (int r = 0; r < 8; ++r) {
                const int m = rowBase + wm * 32 + mi * 16 + half * 8 + r;
                float vv = acc[mi][ni][r];
                if constexpr (BIASF)    vv += bv;
                if constexpr (MODE == 1) vv = fmaxf(vv, 0.0f);
                if constexpr (MODE == 2) {
                    float mi_ = mask[(size_t)b * Sdim + m];
                    vv = expf(tanhf(vv)) * mi_ * mj + EPSC;
                    if (m == n) vv = 0.0f;    // A * (1 - eye)
                }
                size_t off = (size_t)b * strideC + (size_t)m * ldc + n;
                if constexpr (WF) outF[off] = vv;
                if constexpr (WB) outB[off] = (bf16)vv;
            }
        }
    }
}

// ---------------------------------------------------------------------------
// weight convert: fp32 [K][N] -> bf16 transposed [N][K]
// ---------------------------------------------------------------------------
__global__ __launch_bounds__(256) void k_convw(const float* __restrict__ in,
                                               bf16* __restrict__ out, int K, int N)
{
    __shared__ float tile[32][33];
    int n0 = blockIdx.x * 32, k0 = blockIdx.y * 32;
    int tx = threadIdx.x, ty = threadIdx.y;
    #pragma unroll
    for (int q = 0; q < 4; ++q)
        tile[ty + q * 8][tx] = in[(size_t)(k0 + ty + q * 8) * N + n0 + tx];
    __syncthreads();
    #pragma unroll
    for (int q = 0; q < 4; ++q)
        out[(size_t)(n0 + ty + q * 8) * K + k0 + tx] = (bf16)tile[tx][ty + q * 8];
}

// bf16 batched transpose: in [R][C] -> out [C][R], batch = blockIdx.z
__global__ __launch_bounds__(256) void k_transpose_bf(const bf16* __restrict__ in,
                                                      bf16* __restrict__ out, int R, int C)
{
    __shared__ bf16 tile[32][33];
    size_t base = (size_t)blockIdx.z * R * C;
    int c0 = blockIdx.x * 32, r0 = blockIdx.y * 32;
    int tx = threadIdx.x, ty = threadIdx.y;
    #pragma unroll
    for (int q = 0; q < 4; ++q)
        tile[ty + q * 8][tx] = in[base + (size_t)(r0 + ty + q * 8) * C + c0 + tx];
    __syncthreads();
    #pragma unroll
    for (int q = 0; q < 4; ++q)
        out[base + (size_t)(c0 + ty + q * 8) * R + r0 + tx] = tile[tx][ty + q * 8];
}

// h = x * mask (broadcast over H); fp32 + bf16 copies
__global__ __launch_bounds__(256) void k_maskh(const float* __restrict__ x,
                                               const float* __restrict__ mask,
                                               float* __restrict__ hf, bf16* __restrict__ hb)
{
    size_t idx = (size_t)blockIdx.x * 256 + threadIdx.x;
    size_t r = idx >> 10;
    float v = x[idx] * mask[r];
    hf[idx] = v;
    hb[idx] = (bf16)v;
}

// root = tp . fi_w ; f = exp(tanh(root))*m + eps   (block per row)
__global__ __launch_bounds__(256) void k_root(const bf16* __restrict__ tp,
                                              const float* __restrict__ fi_w,
                                              const float* __restrict__ mask,
                                              float* __restrict__ f_buf)
{
    __shared__ float red[256];
    size_t r = blockIdx.x;
    int tid = threadIdx.x;
    float s = 0.0f;
    #pragma unroll
    for (int q = 0; q < 4; ++q) {
        int c = tid + q * 256;
        s += (float)tp[r * Hn + c] * fi_w[c];
    }
    red[tid] = s;
    __syncthreads();
    for (int off = 128; off > 0; off >>= 1) {
        if (tid < off) red[tid] += red[tid + off];
        __syncthreads();
    }
    if (tid == 0)
        f_buf[r] = expf(tanhf(red[0])) * mask[r] + EPSC;
}

// column sums of A over axis-1: cs[b][j] = sum_i A[b][i][j]
__global__ __launch_bounds__(256) void k_colsum(const float* __restrict__ Am,
                                                float* __restrict__ cs)
{
    int b = blockIdx.x, j = threadIdx.x;
    const float* Ab = Am + (size_t)b * Sn * Sn;
    float s = 0.0f;
    for (int i = 0; i < Sn; ++i) s += Ab[(size_t)i * Sn + j];
    cs[(size_t)b * Sn + j] = s;
}

// L_bar: -A + colsum*eye, row0 := f  -> written into Linv buffer (in-place GJ next)
__global__ __launch_bounds__(256) void k_buildL(const float* __restrict__ Am,
                                                const float* __restrict__ cs,
                                                const float* __restrict__ f_buf,
                                                float* __restrict__ L)
{
    size_t idx = (size_t)blockIdx.x * 256 + threadIdx.x;
    int b = (int)(idx >> 16);
    int i = (int)((idx >> 8) & 255);
    int j = (int)(idx & 255);
    float v = -Am[idx] + (i == j ? cs[(size_t)b * Sn + j] : 0.0f);
    if (i == 0) v = f_buf[(size_t)b * Sn + j];
    L[idx] = v;
}

// ---------------------------------------------------------------------------
// In-place Gauss-Jordan inversion of 256x256, one block (1024 thr) per batch.
// Matrix lives in global (L2-resident: 64 x 256KB = 16MB << 192MB L2).
// Thread (chunk=tid>>8, j=tid&255) owns elements (rows chunk*64..+63, col j).
// Pivot row & column-k snapshot staged in LDS each step.
// ---------------------------------------------------------------------------
__global__ __launch_bounds__(1024) void k_gj(float* __restrict__ Linv)
{
    __shared__ float prow[Sn];
    __shared__ float colk[Sn];
    float* Lb = Linv + (size_t)blockIdx.x * Sn * Sn;
    int tid = threadIdx.x;
    int j = tid & 255;
    int chunk = tid >> 8;
    int i0 = chunk * 64;

    for (int k = 0; k < Sn; ++k) {
        __syncthreads();
        float p  = Lb[(size_t)k * Sn + k];
        float ip = 1.0f / p;
        if (tid < Sn) {
            colk[tid] = Lb[(size_t)tid * Sn + k];
            float pv = Lb[(size_t)k * Sn + tid] * ip;
            if (tid == k) pv = ip;
            prow[tid] = pv;
        }
        __syncthreads();
        if (tid < Sn) Lb[(size_t)k * Sn + tid] = prow[tid];
        float pj = prow[j];
        for (int ii = 0; ii < 64; ++ii) {
            int i = i0 + ii;
            if (i == k) continue;
            float fk = colk[i];
            float v  = (j == k) ? (-fk * ip) : (Lb[(size_t)i * Sn + j] - fk * pj);
            Lb[(size_t)i * Sn + j] = v;
        }
        __threadfence();
    }
}

// d0 = f * Linv[:, :, 0] * m
__global__ __launch_bounds__(256) void k_d0(const float* __restrict__ f_buf,
                                            const float* __restrict__ Linv,
                                            const float* __restrict__ mask,
                                            float* __restrict__ d0)
{
    int b = blockIdx.x, i = threadIdx.x;
    size_t r = (size_t)b * Sn + i;
    d0[r] = f_buf[r] * Linv[((size_t)b << 16) + (size_t)i * Sn] * mask[r];
}

// d = (m1*A*diag - m2*A*Linv^T) * m_i * m_j   (+ bf16 copy for the msg GEMM)
__global__ __launch_bounds__(256) void k_d(const float* __restrict__ Am,
                                           const float* __restrict__ Linv,
                                           const float* __restrict__ mask,
                                           float* __restrict__ dOut,
                                           bf16* __restrict__ dBf)
{
    size_t idx = (size_t)blockIdx.x * 256 + threadIdx.x;
    int b = (int)(idx >> 16);
    int i = (int)((idx >> 8) & 255);
    int j = (int)(idx & 255);
    float a = Am[idx];
    const float* Lb = Linv + ((size_t)b << 16);
    float v = 0.0f;
    if (j != 0) v += a * Lb[(size_t)j * Sn + j];
    if (i != 0) v -= a * Lb[(size_t)j * Sn + i];
    v *= mask[(size_t)b * Sn + i] * mask[(size_t)b * Sn + j];
    dOut[idx] = v;
    dBf[idx]  = (bf16)v;
}

// update = d0*msg + (1-d0)*lin ; writes fp32 update and bf16 [update|h] concat rows
__global__ __launch_bounds__(256) void k_update(const float* __restrict__ d0,
                                                const float* __restrict__ msg,
                                                const float* __restrict__ lin,
                                                const bf16* __restrict__ hb,
                                                float* __restrict__ updF,
                                                bf16* __restrict__ cat)
{
    size_t idx = (size_t)blockIdx.x * 256 + threadIdx.x;
    size_t r = idx >> 10;
    int    c = (int)(idx & 1023);
    float d0v = d0[r];
    float u = d0v * msg[idx] + (1.0f - d0v) * lin[idx];
    updF[idx] = u;
    cat[r * 2048 + c]        = (bf16)u;
    cat[r * 2048 + 1024 + c] = hb[idx];
}

// gate = sigmoid(gp); hn = gate*tanh(u) + (1-gate)*h; LayerNorm; write 3 outputs
__global__ __launch_bounds__(256) void k_final(const float* __restrict__ gp,
                                               const float* __restrict__ upd,
                                               const float* __restrict__ lng,
                                               const float* __restrict__ lnb,
                                               float* __restrict__ hf,
                                               bf16* __restrict__ hb,
                                               float* __restrict__ gnn)
{
    __shared__ float red[256];
    size_t r = blockIdx.x;
    int tid = threadIdx.x;
    float hn[4];
    #pragma unroll
    for (int q = 0; q < 4; ++q) {
        size_t idx = r * Hn + tid + q * 256;
        float g = 1.0f / (1.0f + expf(-gp[idx]));
        hn[q] = g * tanhf(upd[idx]) + (1.0f - g) * hf[idx];
    }
    float s = hn[0] + hn[1] + hn[2] + hn[3];
    red[tid] = s;
    __syncthreads();
    for (int off = 128; off > 0; off >>= 1) {
        if (tid < off) red[tid] += red[tid + off];
        __syncthreads();
    }
    float mu = red[0] * (1.0f / 1024.0f);
    __syncthreads();
    float vs = 0.0f;
    #pragma unroll
    for (int q = 0; q < 4; ++q) { float ddd = hn[q] - mu; vs += ddd * ddd; }
    red[tid] = vs;
    __syncthreads();
    for (int off = 128; off > 0; off >>= 1) {
        if (tid < off) red[tid] += red[tid + off];
        __syncthreads();
    }
    float inv = rsqrtf(red[0] * (1.0f / 1024.0f) + LNE);
    #pragma unroll
    for (int q = 0; q < 4; ++q) {
        int c = tid + q * 256;
        size_t idx = r * Hn + c;
        float o = (hn[q] - mu) * inv * lng[c] + lnb[c];
        hf[idx]  = o;
        hb[idx]  = (bf16)o;
        gnn[idx] = o;
    }
}

// ---------------------------------------------------------------------------
// host side
// ---------------------------------------------------------------------------
template<int MODE, bool BIASF, bool WF, bool WB>
static inline void launch_gemm(hipStream_t s,
                               const bf16* A, int lda, size_t sA,
                               const bf16* Bt, int ldb, size_t sB,
                               const float* bias, float* outF, bf16* outB,
                               int ldc, size_t sC, int M, int N, int K, int batch,
                               const float* mask, int Sdim)
{
    dim3 grid(N / BN, M / BM, batch);
    k_gemm<MODE, BIASF, WF, WB><<<grid, 256, 0, s>>>(
        A, lda, sA, Bt, ldb, sB, bias, outF, outB, ldc, sC, K, mask, Sdim);
}

#define MB (1ULL << 20)

extern "C" void kernel_launch(void* const* d_in, const int* in_sizes, int n_in,
                              void* d_out, int out_size, void* d_ws, size_t ws_size,
                              hipStream_t stream)
{
    const float* x     = (const float*)d_in[0];
    const float* mask  = (const float*)d_in[1];
    const float* tp_w  = (const float*)d_in[2];
    const float* tp_b  = (const float*)d_in[3];
    const float* tc_w  = (const float*)d_in[4];
    const float* tc_b  = (const float*)d_in[5];
    const float* bil_w = (const float*)d_in[6];
    const float* fi_w  = (const float*)d_in[7];
    const float* wi_w  = (const float*)d_in[8];
    const float* wi_b  = (const float*)d_in[9];
    const float* wg_w  = (const float*)d_in[10];
    const float* wg_b  = (const float*)d_in[11];
    const float* ln_g  = (const float*)d_in[12];
    const float* ln_b  = (const float*)d_in[13];

    float* out   = (float*)d_out;
    float* gnn   = out;                                   // [2,B,S,H]
    float* dOut  = out + (size_t)2 * Bn * Sn * Hn;        // [B,S,S]
    float* d0Out = dOut + (size_t)Bn * Sn * Sn;           // [B,S]

    char* ws = (char*)d_ws;
    // phase-aliased layout (~475 MiB total)
    float* h_f   = (float*)(ws + 0 * MB);        // 64MB
    bf16*  h_b   = (bf16*)(ws + 64 * MB);        // 32MB
    bf16*  ht_b  = (bf16*)(ws + 96 * MB);        // 32MB
    bf16*  tp_bf = (bf16*)(ws + 128 * MB);       // 32MB   (dead by hop phase)
    bf16*  tc_bf = (bf16*)(ws + 160 * MB);       // 32MB   (dead by hop phase)
    bf16*  tpb_b = (bf16*)(ws + 192 * MB);       // 32MB   (dead by hop phase)
    float* A_mat = (float*)(ws + 224 * MB);      // 16MB   (dead by hop phase)
    float* Linv  = (float*)(ws + 240 * MB);      // 16MB   (dead by hop phase)
    float* msg_f = (float*)(ws + 128 * MB);      // 64MB   (hop phase, aliases tp/tc)
    float* lin_f = (float*)(ws + 192 * MB);      // 64MB   (hop phase, aliases tpb/A/Linv)
    float* upd_f = (float*)(ws + 256 * MB);      // 64MB
    bf16*  cat_b = (bf16*)(ws + 320 * MB);       // 64MB  [M][2H]
    float* gat_f = (float*)(ws + 384 * MB);      // 64MB
    bf16*  d_bf  = (bf16*)(ws + 448 * MB);       // 8MB
    bf16*  tpwT  = (bf16*)(ws + 456 * MB);       // 2MB each
    bf16*  tcwT  = (bf16*)(ws + 458 * MB);
    bf16*  bilT  = (bf16*)(ws + 460 * MB);
    bf16*  wiT0  = (bf16*)(ws + 462 * MB);
    bf16*  wiT1  = (bf16*)(ws + 464 * MB);
    bf16*  wgT0  = (bf16*)(ws + 466 * MB);       // 4MB
    bf16*  wgT1  = (bf16*)(ws + 470 * MB);       // 4MB
    float* cs    = (float*)(ws + 474 * MB);      // 64KB
    float* f_buf = (float*)(ws + 474 * MB + 65536);

    const size_t MSH = (size_t)Bn * Sn * Hn;     // 16,777,216
    const size_t MSS = (size_t)Bn * Sn * Sn;     //  4,194,304
    dim3 tb(32, 8);

    // --- weight conversions (fp32 [K][N] -> bf16 [N][K]) ---
    k_convw<<<dim3(32, 32), tb, 0, stream>>>(tp_w, tpwT, Hn, Hn);
    k_convw<<<dim3(32, 32), tb, 0, stream>>>(tc_w, tcwT, Hn, Hn);
    k_convw<<<dim3(32, 32), tb, 0, stream>>>(bil_w, bilT, Hn, Hn);
    k_convw<<<dim3(32, 32), tb, 0, stream>>>(wi_w,           wiT0, Hn, Hn);
    k_convw<<<dim3(32, 32), tb, 0, stream>>>(wi_w + Hn * Hn, wiT1, Hn, Hn);
    k_convw<<<dim3(32, 64), tb, 0, stream>>>(wg_w,               wgT0, 2 * Hn, Hn);
    k_convw<<<dim3(32, 64), tb, 0, stream>>>(wg_w + 2 * Hn * Hn, wgT1, 2 * Hn, Hn);

    // --- h = x * m ---
    k_maskh<<<MSH / 256, 256, 0, stream>>>(x, mask, h_f, h_b);

    // --- tp / tc / tpb GEMMs (WMMA) ---
    launch_gemm<1, true, false, true>(stream, h_b, Hn, 0, tpwT, Hn, 0, tp_b,
                                      nullptr, tp_bf, Hn, 0, Bn * Sn, Hn, Hn, 1, nullptr, 0);
    launch_gemm<1, true, false, true>(stream, h_b, Hn, 0, tcwT, Hn, 0, tc_b,
                                      nullptr, tc_bf, Hn, 0, Bn * Sn, Hn, Hn, 1, nullptr, 0);
    launch_gemm<0, false, false, true>(stream, tp_bf, Hn, 0, bilT, Hn, 0, nullptr,
                                       nullptr, tpb_b, Hn, 0, Bn * Sn, Hn, Hn, 1, nullptr, 0);

    // --- root -> f ---
    k_root<<<Bn * Sn, 256, 0, stream>>>(tp_bf, fi_w, mask, f_buf);

    // --- scores + A-matrix epilogue:  A = exp(tanh(tpb @ tc^T))*mm+eps, diag 0 ---
    launch_gemm<2, false, true, false>(stream, tpb_b, Hn, (size_t)Sn * Hn,
                                       tc_bf, Hn, (size_t)Sn * Hn,
                                       nullptr, A_mat, nullptr, Sn, (size_t)Sn * Sn,
                                       Sn, Sn, Hn, Bn, mask, Sn);

    // --- Laplacian + inversion ---
    k_colsum<<<Bn, 256, 0, stream>>>(A_mat, cs);
    k_buildL<<<MSS / 256, 256, 0, stream>>>(A_mat, cs, f_buf, Linv);
    k_gj<<<Bn, 1024, 0, stream>>>(Linv);

    // --- marginals d0, d ---
    k_d0<<<Bn, 256, 0, stream>>>(f_buf, Linv, mask, d0Out);
    k_d<<<MSS / 256, 256, 0, stream>>>(A_mat, Linv, mask, dOut, d_bf);

    // --- GNN hops ---
    const bf16* wiT[2] = { wiT0, wiT1 };
    const bf16* wgT[2] = { wgT0, wgT1 };
    for (int hop = 0; hop < 2; ++hop) {
        // h^T per batch: [S][H] -> [H][S]
        k_transpose_bf<<<dim3(Hn / 32, Sn / 32, Bn), tb, 0, stream>>>(h_b, ht_b, Sn, Hn);
        // msg = d @ h  (batched, K=S)
        launch_gemm<0, false, true, false>(stream, d_bf, Sn, (size_t)Sn * Sn,
                                           ht_b, Sn, (size_t)Hn * Sn,
                                           nullptr, msg_f, nullptr, Hn, (size_t)Sn * Hn,
                                           Sn, Hn, Sn, Bn, nullptr, 0);
        // lin = h @ wi + b
        launch_gemm<0, true, true, false>(stream, h_b, Hn, 0, wiT[hop], Hn, 0,
                                          wi_b + hop * Hn, lin_f, nullptr, Hn, 0,
                                          Bn * Sn, Hn, Hn, 1, nullptr, 0);
        // update + concat rows
        k_update<<<MSH / 256, 256, 0, stream>>>(d0Out, msg_f, lin_f, h_b, upd_f, cat_b);
        // gate_pre = [update|h] @ wg + b   (K = 2H)
        launch_gemm<0, true, true, false>(stream, cat_b, 2 * Hn, 0, wgT[hop], 2 * Hn, 0,
                                          wg_b + hop * Hn, gat_f, nullptr, Hn, 0,
                                          Bn * Sn, Hn, 2 * Hn, 1, nullptr, 0);
        // gate/tanh/LN fused; writes next h (fp32+bf16) and gnn_out[hop]
        k_final<<<Bn * Sn, 256, 0, stream>>>(gat_f, upd_f,
                                             ln_g + hop * Hn, ln_b + hop * Hn,
                                             h_f, h_b, gnn + (size_t)hop * MSH);
    }
    (void)in_sizes; (void)n_in; (void)out_size; (void)ws_size;
}
